// HardBinaryConv_54125177864368
// MI455X (gfx1250) — matrix operand order repaired and checked by reading, not compiled, not run on hardware
//
#include <hip/hip_runtime.h>

// ---------------------------------------------------------------------------
// HardBinaryConv on gfx1250: implicit-GEMM with v_wmma_f32_16x16x32_f16.
//   M = OUT(256), N = B*H*W(100352), K = IN*3*3(2304)
//   Wave tile: 16(M) x 64(N) via 4 accumulators -> 288 WMMA/wave.
//   Double-buffered pipeline: global_load_async_to_lds_b128 stages the next
//   64px x 128ch half-slab while WMMAs consume the current one (ASYNCcnt).
// ---------------------------------------------------------------------------

#define B_   32
#define CIN  256
#define COUT 256
#define HH   56
#define WW   56
#define HW   (HH * WW)            // 3136
#define NPOS (B_ * HW)            // 100352
#define KFLAT (CIN * 9)           // 2304
#define NPIX 64                   // pixels per workgroup
#define HCH  128                  // channels per pipeline step (half of CIN)
#define NSTEP (9 * 2)             // 9 taps x 2 channel halves

typedef __attribute__((ext_vector_type(16))) _Float16 v16h;
typedef __attribute__((ext_vector_type(8)))  float    v8f;

union Frag16 { v16h v; uint4 q[2]; };

__device__ __forceinline__ unsigned short f32_to_h(float f) {
  union { _Float16 h; unsigned short u; } c;
  c.h = (_Float16)f;              // RNE conversion
  return c.u;
}

// ---------------------------------------------------------------------------
// Kernel 1: per-output-channel scale = mean|w|, and sign(w) packed as f16
// into Wp[(r*256 + o)*256 + ci]  (r = kh*3+kw), so A-fragments are contiguous.
// ---------------------------------------------------------------------------
__global__ __launch_bounds__(256) void prep_weights_kernel(
    const float* __restrict__ w,           // [O][I][3][3] flat
    unsigned short* __restrict__ Wp,       // [9][O][I] f16 sign
    float* __restrict__ scale)             // [O]
{
  __shared__ float red[256];
  const int o = blockIdx.x;                // 0..255
  const int t = threadIdx.x;
  float s = 0.0f;
#pragma unroll
  for (int j = 0; j < 9; ++j) {
    const int idx = t + j * 256;           // 0..2303 = ci*9 + r
    const float v = w[(size_t)o * KFLAT + idx];
    s += fabsf(v);
    const int ci = idx / 9;
    const int r  = idx - ci * 9;
    unsigned short us = 0;
    if (v > 0.0f) us = 0x3C00;             // +1.0h
    else if (v < 0.0f) us = 0xBC00;        // -1.0h   (sign(0) == 0)
    Wp[((size_t)(r * COUT + o) << 8) + ci] = us;
  }
  red[t] = s;
  __syncthreads();
  for (int off = 128; off > 0; off >>= 1) {
    if (t < off) red[t] += red[t + off];
    __syncthreads();
  }
  if (t == 0) scale[o] = red[0] * (1.0f / (float)KFLAT);
}

// ---------------------------------------------------------------------------
// Kernel 2: x f32 NCHW -> f16 NHWC via 32(ch) x 32(pix) LDS transpose tile.
// Coalesced f32 reads, coalesced 8B f16 writes.
// ---------------------------------------------------------------------------
__global__ __launch_bounds__(256) void x_to_f16_nhwc_kernel(
    const float* __restrict__ x,           // [B][C][H][W]
    unsigned short* __restrict__ X16)      // [B][H][W][C] f16
{
  __shared__ float tile[32][33];
  const int t  = threadIdx.x;
  const int pt = blockIdx.x;               // 0..3135 : pixel tile (32 pixels)
  const int c0 = blockIdx.y * 32;          // channel tile
  const int b  = pt / (HW / 32);           // HW/32 = 98, exact
  const int p0 = (pt % (HW / 32)) * 32;
  const size_t ibase = (size_t)b * CIN * HW;

  const int pix = t & 31, cl = t >> 5;     // 32 pixels x 8 channel rows
#pragma unroll
  for (int j = 0; j < 4; ++j) {
    const int c = cl + j * 8;              // 0..31
    tile[c][pix] = x[ibase + (size_t)(c0 + c) * HW + p0 + pix];
  }
  __syncthreads();

  const int cch = t & 7, pl = t >> 3;      // 8 chunks of 4 ch x 32 pixels
  ushort4 o4;
  o4.x = f32_to_h(tile[cch * 4 + 0][pl]);
  o4.y = f32_to_h(tile[cch * 4 + 1][pl]);
  o4.z = f32_to_h(tile[cch * 4 + 2][pl]);
  o4.w = f32_to_h(tile[cch * 4 + 3][pl]);
  *(ushort4*)(X16 + (((size_t)(b * HW + p0 + pl)) << 8) + c0 + cch * 4) = o4;
}

// ---------------------------------------------------------------------------
// Kernel 3: WMMA conv. Workgroup = 8 waves = 128 out-ch x 64 pixels.
// 18-step pipeline over (tap, ci-half): async-stage 16 KB half-slab for
// step s+1 while computing 4 ci-steps x 4 N-subtiles = 16 WMMAs for step s.
// ---------------------------------------------------------------------------
__global__ __launch_bounds__(256) void conv_wmma_kernel(
    const unsigned short* __restrict__ Wp,   // [9][O][I] f16 sign
    const unsigned short* __restrict__ X16,  // [B][H][W][C] f16
    const float* __restrict__ scale,         // [O]
    float* __restrict__ out)                 // [B][O][H][W] f32
{
  __shared__ unsigned short Bs[2][NPIX * HCH]; // 2 x 16 KB half-slabs
  __shared__ int ph[NPIX], pw[NPIX], pb[NPIX];

  const int t    = threadIdx.x;
  const int lane = t & 31;
  const int wave = t >> 5;
  const int n0   = blockIdx.x * NPIX;               // pixel tile
  const int o_base = blockIdx.y * 128 + wave * 16;  // out-channel tile

  if (t < NPIX) {
    const int g = n0 + t;
    const int b = g / HW, rem = g % HW;
    pb[t] = b;
    ph[t] = rem / WW;
    pw[t] = rem % WW;
  }
  __syncthreads();

  // staging role: 4 threads per pixel, 64 bytes (32 f16 channels) each
  const int pm  = t >> 2;                 // pixel 0..63
  const int seg = t & 3;                  // 64B segment within 256B half-row
  const int sb = pb[pm], sh = ph[pm], sw = pw[pm];
  // LDS byte addresses for this thread's staging chunk (generic addr[31:0]
  // is the LDS offset per the flat->LDS aperture mapping)
  const unsigned lds_dst0 = (unsigned)(size_t)(&Bs[0][pm * HCH + seg * 32]);
  const unsigned lds_dst1 = (unsigned)(size_t)(&Bs[1][pm * HCH + seg * 32]);
  const size_t gpix = ((size_t)(sb * HW) << 8) + seg * 32;

  // issue the async (or zero-fill) staging for pipeline step `step`
  auto stage = [&](int step) {
    const int r    = step >> 1;
    const int half = step & 1;
    const int hh = sh + (r / 3 - 1);
    const int ww = sw + (r % 3 - 1);
    const unsigned ldst = (step & 1) ? lds_dst1 : lds_dst0;
    if (hh >= 0 && hh < HH && ww >= 0 && ww < WW) {
      const unsigned short* gsrc =
          X16 + gpix + (((size_t)(hh * WW + ww)) << 8) + half * HCH;
#pragma unroll
      for (int j = 0; j < 4; ++j) {
        asm volatile("global_load_async_to_lds_b128 %0, %1, off"
                     :: "v"(ldst + j * 16), "v"(gsrc + j * 8)
                     : "memory");
      }
    } else {
      const uint4 z = {0, 0, 0, 0};
      uint4* dst = (uint4*)(&Bs[step & 1][pm * HCH + seg * 32]);
#pragma unroll
      for (int j = 0; j < 4; ++j) dst[j] = z;
    }
  };

  // WMMA fragment roles (wave32, 16-bit A / B layouts)
  const int m   = lane & 15;              // A row / B col / D col
  const int kba = (lane >> 4) << 3;       // A k-base: 0 or 8
  const int kbb = (lane >> 4) << 4;       // unused (B also uses kba): 0 or 16

  v8f acc[4] = {};

  // prologue: fill buffer 0 with (tap 0, half 0)
  stage(0);
  asm volatile("s_wait_asynccnt 0x0" ::: "memory");
  __syncthreads();

  for (int step = 0; step < NSTEP; ++step) {
    const int r    = step >> 1;
    const int half = step & 1;

    // kick off DMA for the next half-slab into the other buffer
    if (step + 1 < NSTEP) stage(step + 1);

    const unsigned short* arow =
        Wp + (((size_t)(r * COUT + o_base + m)) << 8) + half * HCH + kba;
    if (half == 0 && r < 8)
      __builtin_prefetch(arow + (size_t)COUT * CIN, 0, 0);

    const unsigned short* brow = &Bs[half ^ (r & 1) ^ half][0]; // silence
    (void)brow;
    const unsigned short* bbase = &Bs[step & 1][0];

#pragma unroll
    for (int ci0 = 0; ci0 < HCH; ci0 += 32) {
      Frag16 a;
      // A: lane m holds K = ci0+kba+{0..7} and ci0+kba+16+{0..7}
      const uint4* ap = (const uint4*)(arow + ci0);
      a.q[0] = ap[0];
      a.q[1] = ap[2];
#pragma unroll
      for (int s2 = 0; s2 < 4; ++s2) {
        Frag16 b;
        // B: lane n holds K = ci0+kba+{0..7} and ci0+kba+16+{0..7}
        const uint4* bp =
            (const uint4*)(bbase + (s2 * 16 + m) * HCH + ci0 + kba);
        b.q[0] = bp[0];
        b.q[1] = bp[2];
        acc[s2] = __builtin_amdgcn_wmma_f32_16x16x32_f16(
            /*neg_a=*/false, a.v, /*neg_b=*/false, b.v,
            /*c_mod=*/(short)0, acc[s2], /*reuse_a=*/false, /*reuse_b=*/false);
      }
    }

    // publish the next buffer: our DMA must be done, everyone must be done
    asm volatile("s_wait_asynccnt 0x0" ::: "memory");
    __syncthreads();
  }

  // ---- scaled write-out: D f32 16x16, lane n = lane&15, VGPR i -> m=i+kba
#pragma unroll
  for (int s2 = 0; s2 < 4; ++s2) {
    const int n = (lane & 15) + s2 * 16;
    const size_t obase =
        (size_t)pb[n] * COUT * HW + (size_t)ph[n] * WW + pw[n];
#pragma unroll
    for (int i = 0; i < 8; ++i) {
      const int o = o_base + kba + i;
      out[obase + (size_t)o * HW] = acc[s2][i] * scale[o];
    }
  }
}

// ---------------------------------------------------------------------------
extern "C" void kernel_launch(void* const* d_in, const int* in_sizes, int n_in,
                              void* d_out, int out_size, void* d_ws, size_t ws_size,
                              hipStream_t stream) {
  const float* x   = (const float*)d_in[0];   // [32][256][56][56] f32
  const float* wts = (const float*)d_in[1];   // [256*256*9] f32
  float* out = (float*)d_out;                 // [32][256][56][56] f32

  // workspace layout (256B-aligned sections)
  char* ws = (char*)d_ws;
  float* scale = (float*)ws;                                    // 1 KB
  unsigned short* Wp  = (unsigned short*)(ws + 1024);           // 1.18 MB
  unsigned short* X16 = (unsigned short*)(ws + 1024 + (size_t)9 * COUT * CIN * 2);

  prep_weights_kernel<<<COUT, 256, 0, stream>>>(wts, Wp, scale);
  x_to_f16_nhwc_kernel<<<dim3(B_ * (HW / 32), CIN / 32), 256, 0, stream>>>(x, X16);
  conv_wmma_kernel<<<dim3(NPOS / NPIX, COUT / 128), 256, 0, stream>>>(Wp, X16, scale, out);
}
